// BahdanauAttention_70274254897902
// MI455X (gfx1250) — matrix operand order repaired and checked by reading, not compiled, not run on hardware
//
#include <hip/hip_runtime.h>
#include <hip/hip_bf16.h>

typedef __attribute__((ext_vector_type(2)))  float        v2f;
typedef __attribute__((ext_vector_type(8)))  float        v8f;
typedef __attribute__((ext_vector_type(16))) __bf16       v16bf;
typedef __attribute__((ext_vector_type(4)))  unsigned int u32x4;
typedef __attribute__((ext_vector_type(8)))  int          i32x8;
typedef __attribute__((ext_vector_type(4)))  int          i32x4;

#define B_   4
#define S_   512
#define D_   256
#define U_   128
#define NEG_ (-1e9f)

#define SCORE_STRIDE 532   // 532 % 64 == 20; 20*r mod 64 distinct for r=0..15
#define POOL_STRIDE  132   // 128 data + 4 pad DWORDs (TDM pad feature)

__device__ __forceinline__ float tanh_fast(float x) {
#if __has_builtin(__builtin_amdgcn_tanhf)
  return __builtin_amdgcn_tanhf(x);   // v_tanh_f32 on gfx1250
#else
  return tanhf(x);
#endif
}

// ---------------------------------------------------------------------------
// Phase 1: Q = values @ Wq, Vp = values @ Wv  (fp32 WMMA, full precision)
// ---------------------------------------------------------------------------
__global__ __launch_bounds__(32)
void proj_kernel(const float* __restrict__ values,
                 const float* __restrict__ Wq,
                 const float* __restrict__ Wv,
                 float* __restrict__ Q,
                 float* __restrict__ Vp) {
  const int l    = threadIdx.x;
  const int m    = l & 15;
  const int hi   = l >> 4;
  const int row0 = blockIdx.x * 16;    // over B*S rows
  const int col0 = blockIdx.y * 16;    // over U cols
  const float* W   = blockIdx.z ? Wv : Wq;
  float*       Out = blockIdx.z ? Vp  : Q;

  const float* arow = values + (size_t)(row0 + m) * D_;
  v8f acc = {};
  #pragma unroll 4
  for (int k = 0; k < D_; k += 4) {
    const int ka = k + hi * 2;
    v2f a, bm;
    a.x  = arow[ka];
    a.y  = arow[ka + 1];
    bm.x = W[(size_t)ka       * U_ + col0 + m];
    bm.y = W[(size_t)(ka + 1) * U_ + col0 + m];
    acc = __builtin_amdgcn_wmma_f32_16x16x4_f32(false, a, false, bm,
                                                (short)0, acc, false, false);
  }
  #pragma unroll
  for (int r = 0; r < 8; ++r)
    Out[(size_t)(row0 + r + 8 * hi) * U_ + col0 + m] = acc[r];
}

// ---------------------------------------------------------------------------
// Phase 2: fused score(tanh) + causal softmax + context WMMA.
// One 256-thread block per (batch, 16-query-row tile).
// Vp chunks are staged into LDS by the Tensor Data Mover (TDM) with the
// pad feature generating 132-DWORD rows directly; fenced on TENSORcnt.
// ---------------------------------------------------------------------------
__global__ __launch_bounds__(256)
void attn_kernel(const float* __restrict__ values,
                 const float* __restrict__ Q,
                 const float* __restrict__ Vp,
                 const float* __restrict__ Vw,
                 float* __restrict__ out) {
  __shared__ float sQ[16 * POOL_STRIDE];        // Q tile, padded rows
  __shared__ float sVw[U_];
  __shared__ float sScore[16 * SCORE_STRIDE];   // fp32 scores -> exp in place
  __shared__ float sPool[32 * POOL_STRIDE];     // TDM dest; later: bf16 attn
  __shared__ float sRed[16 * 17];
  __shared__ float sRowMax[16];
  __shared__ float sRowSum[16];

  const int tid = threadIdx.x;
  const int bz  = blockIdx.y;          // batch
  const int i0  = blockIdx.x * 16;     // query-row tile within batch

  // ---- stage Q tile and Vw ----
  for (int t = tid; t < 16 * U_; t += 256) {
    const int r = t >> 7, u = t & 127;
    sQ[r * POOL_STRIDE + u] = Q[((size_t)bz * S_ + i0 + r) * U_ + u];
  }
  if (tid < U_) sVw[tid] = Vw[tid];
  __syncthreads();

  // ---- score phase ----
  for (int jc = 0; jc < S_; jc += 32) {
    const float* src = Vp + ((size_t)bz * S_ + jc) * U_;
#if __has_builtin(__builtin_amdgcn_tensor_load_to_lds) && __has_builtin(__builtin_amdgcn_s_wait_tensorcnt)
    if (tid < 32) {                    // wave 0 drives the TDM
      const unsigned long long ga = (unsigned long long)(uintptr_t)src;
      const unsigned lds = (unsigned)(uintptr_t)&sPool[0];  // generic[31:0] = LDS offset
      u32x4 g0;
      g0[0] = 1u;                                  // count=1, user descriptor
      g0[1] = lds;                                 // lds_addr
      g0[2] = (unsigned)(ga & 0xffffffffu);        // global_addr[31:0]
      g0[3] = (unsigned)((ga >> 32) & 0x01ffffffu) // global_addr[56:32]
              | (2u << 30);                        // type=2 ("image")
      i32x8 g1;
      g1[0] = (2 << 16)        // data_size = 4B
            | (1 << 20)        // pad_enable
            | (6 << 22)        // pad_interval: 128 DWORDs
            | (3 << 25);       // pad_amount:   4 DWORDs -> 132-DWORD LDS rows
      g1[1] = (U_ & 0xffff) << 16;                 // tensor_dim0 = 128
      g1[2] = (S_ & 0xffff) << 16;                 // tensor_dim1 = 512
      g1[3] = (U_ & 0xffff) << 16;                 // tile_dim0 = 128
      g1[4] = 32;                                  // tile_dim1 = 32 rows
      g1[5] = U_;                                  // tensor_dim0_stride = 128
      g1[6] = 0;
      g1[7] = 0;
      i32x4 gz4 = {0, 0, 0, 0};
      i32x8 gz8 = {0, 0, 0, 0, 0, 0, 0, 0};
      __builtin_amdgcn_tensor_load_to_lds(g0, g1, gz4, gz4, gz8, 0);
      __builtin_amdgcn_s_wait_tensorcnt(0);
    }
#else
    for (int t = tid; t < 32 * U_; t += 256) {
      const int jj = t >> 7, u = t & 127;
      sPool[jj * POOL_STRIDE + u] = src[(size_t)jj * U_ + u];
    }
#endif
    __syncthreads();
    for (int p = tid; p < 512; p += 256) {       // 2 (row,j) pairs per thread
      const int r  = p & 15;
      const int jj = p >> 4;
      const float* qr = &sQ[r * POOL_STRIDE];
      const float* vr = &sPool[jj * POOL_STRIDE];
      float s = 0.f;
      #pragma unroll 8
      for (int u = 0; u < U_; ++u)
        s += sVw[u] * tanh_fast(qr[u] + vr[u]);
      const int j = jc + jj;
      if (j > i0 + r) s += NEG_;                 // causal additive mask
      sScore[r * SCORE_STRIDE + j] = s;
    }
    __syncthreads();
  }

  // ---- causal softmax: 16 threads per row ----
  {
    const int r = tid >> 4;
    const int c = tid & 15;
    float mx = -3.4e38f;
    for (int j = c; j < S_; j += 16) mx = fmaxf(mx, sScore[r * SCORE_STRIDE + j]);
    sRed[r * 17 + c] = mx;
    __syncthreads();
    if (tid < 16) {
      float m2 = sRed[tid * 17];
      #pragma unroll
      for (int k = 1; k < 16; ++k) m2 = fmaxf(m2, sRed[tid * 17 + k]);
      sRowMax[tid] = m2;
    }
    __syncthreads();
    const float rm = sRowMax[r];
    float sum = 0.f;
    for (int j = c; j < S_; j += 16) {
      const float e = __expf(sScore[r * SCORE_STRIDE + j] - rm);
      sScore[r * SCORE_STRIDE + j] = e;
      sum += e;
    }
    sRed[r * 17 + c] = sum;
    __syncthreads();
    if (tid < 16) {
      float s2 = 0.f;
      #pragma unroll
      for (int k = 0; k < 16; ++k) s2 += sRed[tid * 17 + k];
      sRowSum[tid] = s2;
    }
    __syncthreads();
    const float inv = 1.f / sRowSum[r];
    __bf16* attn = (__bf16*)sPool;               // reuse pool (16*524*2 <= pool)
    for (int j = c; j < S_; j += 16)
      attn[r * 524 + j] = (__bf16)(sScore[r * SCORE_STRIDE + j] * inv);
    __syncthreads();
  }

  // ---- context phase: out_tile = attn[16xS] @ values[SxD], bf16 WMMA ----
  {
    const __bf16* attn = (const __bf16*)sPool;
    const int l  = tid & 31;
    const int w  = tid >> 5;                     // wave id 0..7
    const int m  = l & 15;
    const int hi = l >> 4;
    const float* vb = values + (size_t)bz * S_ * D_;
    for (int nt = w; nt < D_ / 16; nt += 8) {    // 2 column tiles per wave
      const int n0 = nt * 16;
      v8f acc = {};
      #pragma unroll 2
      for (int kk = 0; kk < S_; kk += 32) {
        v16bf a, bm;
        #pragma unroll
        for (int e = 0; e < 16; ++e) {
          const int ka = (e < 8 ? e : e + 8) + (hi ? 8 : 0);   // A 16x32 layout
          a[e] = attn[m * 524 + kk + ka];
          const int kb = (e < 8 ? e : e + 8) + (hi ? 8 : 0);   // B 32x16 layout
          bm[e] = (__bf16)vb[(size_t)(kk + kb) * D_ + n0 + m];
        }
        acc = __builtin_amdgcn_wmma_f32_16x16x32_bf16(false, a, false, bm,
                                                      (short)0, acc, false, false);
      }
      #pragma unroll
      for (int r = 0; r < 8; ++r)
        out[((size_t)bz * S_ + i0 + r + 8 * hi) * D_ + n0 + m] = acc[r];
    }
  }
}

// ---------------------------------------------------------------------------
extern "C" void kernel_launch(void* const* d_in, const int* in_sizes, int n_in,
                              void* d_out, int out_size, void* d_ws, size_t ws_size,
                              hipStream_t stream) {
  const float* values = (const float*)d_in[0];   // [B,S,D]
  const float* Wq     = (const float*)d_in[1];   // [D,U]
  const float* Wv     = (const float*)d_in[2];   // [D,U]
  const float* Vw     = (const float*)d_in[3];   // [U]
  float* out = (float*)d_out;                    // [B,S,D]

  float* Q  = (float*)d_ws;                      // [B,S,U] fp32
  float* Vp = Q + (size_t)B_ * S_ * U_;          // [B,S,U] fp32

  dim3 g1((B_ * S_) / 16, U_ / 16, 2);
  proj_kernel<<<g1, 32, 0, stream>>>(values, Wq, Wv, Q, Vp);

  dim3 g2(S_ / 16, B_);
  attn_kernel<<<g2, 256, 0, stream>>>(values, Q, Vp, Vw, out);
}
